// HiSCN_9775345566118
// MI455X (gfx1250) — matrix operand order repaired and checked by reading, not compiled, not run on hardware
//
#include <hip/hip_runtime.h>
#include <hip/hip_bf16.h>

typedef __attribute__((ext_vector_type(16))) __bf16 v16bf;
typedef __attribute__((ext_vector_type(8)))  float  v8f;

#define N_NODES 100000
#define E_EDGES 3200000
#define F_INF   512
#define HID     128
#define KHOPS   10
#define NCLS    10

// ---------------------------------------------------------------------------
// Pack W_in (f32 [2,512,128]) into the WMMA 16-bit B-matrix lane layout:
// for branch br, k-step t (K=32 chunk), ntile (16 cols), lane, elem e:
//   lanes 0-15  hold K = t*32 + {0..7, 16..23} of column n0+lane
//   lanes 16-31 hold K = t*32 + {8..15, 24..31} of column n0+(lane-16)
// ---------------------------------------------------------------------------
__global__ void __launch_bounds__(256)
pack_w_kernel(const float* __restrict__ W_in, __bf16* __restrict__ wpack) {
    int idx = blockIdx.x * 256 + threadIdx.x;        // 2*16*8*32*16 = 131072
    int e     =  idx        & 15;
    int lane  = (idx >> 4)  & 31;
    int ntile = (idx >> 9)  & 7;
    int t     = (idx >> 12) & 15;
    int br    =  idx >> 16;
    int k = t * 32 + ((lane >> 4) * 8) + (e & 7) + ((e >> 3) * 16);
    int n = ntile * 16 + (lane & 15);
    wpack[idx] = (__bf16)W_in[((size_t)br * F_INF + k) * HID + n];
}

// ---------------------------------------------------------------------------
// h = x @ W_in[branch] + b_in[branch];  hidden = fW[branch,0] * h
// Block = 256 threads = 8 waves; block owns a 16-row M tile, wave w owns
// N columns [16w, 16w+16). K loop: 16 steps of 32 (bf16 WMMA, f32 acc).
// ---------------------------------------------------------------------------
__global__ void __launch_bounds__(256)
gemm_in_kernel(const float* __restrict__ x, const __bf16* __restrict__ wpack,
               const float* __restrict__ b_in, const float* __restrict__ fW,
               int branch, float* __restrict__ h, float* __restrict__ hidden) {
    const int lane = threadIdx.x & 31;
    const int wave = threadIdx.x >> 5;               // ntile 0..7
    const int m0   = blockIdx.x * 16;
    const int rowA = m0 + (lane & 15);
    const int koffA = (lane >> 4) * 8;               // A layout: 0 or 8
    const float* xrow = x + (size_t)rowA * F_INF + koffA;

    v8f c = {};
#pragma unroll
    for (int t = 0; t < 16; ++t) {
        const float* xp = xrow + t * 32;
        float4 f0 = *(const float4*)(xp);            // K = koffA + 0..3
        float4 f1 = *(const float4*)(xp + 4);        // K = koffA + 4..7
        float4 f2 = *(const float4*)(xp + 16);       // K = 16+koffA + 0..3
        float4 f3 = *(const float4*)(xp + 20);       // K = 16+koffA + 4..7
        v16bf a;
        a[0]  = (__bf16)f0.x; a[1]  = (__bf16)f0.y; a[2]  = (__bf16)f0.z; a[3]  = (__bf16)f0.w;
        a[4]  = (__bf16)f1.x; a[5]  = (__bf16)f1.y; a[6]  = (__bf16)f1.z; a[7]  = (__bf16)f1.w;
        a[8]  = (__bf16)f2.x; a[9]  = (__bf16)f2.y; a[10] = (__bf16)f2.z; a[11] = (__bf16)f2.w;
        a[12] = (__bf16)f3.x; a[13] = (__bf16)f3.y; a[14] = (__bf16)f3.z; a[15] = (__bf16)f3.w;
        v16bf b = *(const v16bf*)(wpack + (((size_t)branch * 16 + t) * 8 + wave) * 512
                                        + (size_t)lane * 16);
        c = __builtin_amdgcn_wmma_f32_16x16x32_bf16(
                false, a, false, b, (short)0, c, false, false);
    }

    const float fw0  = fW[branch * (KHOPS + 1)];
    const int   col  = wave * 16 + (lane & 15);
    const float bias = b_in[branch * HID + col];
    const int   mb   = m0 + ((lane >> 4) ? 8 : 0);   // C/D layout: VGPR r -> M = mb+r
#pragma unroll
    for (int r = 0; r < 8; ++r) {
        float v = c[r] + bias;
        size_t o = (size_t)(mb + r) * HID + col;
        h[o] = v;
        hidden[o] = fw0 * v;
    }
}

// ---------------------------------------------------------------------------
// COO SpMM: hout[row] += w * hin[col].  One wave per edge, lane owns float4.
// hin/hout are 51.2 MB each -> L2-resident; fp32 atomics resolve in L2.
// ---------------------------------------------------------------------------
__global__ void __launch_bounds__(256)
spmm_kernel(const int* __restrict__ ei, const float* __restrict__ ew,
            const float* __restrict__ hin, float* __restrict__ hout) {
    const int lane = threadIdx.x & 31;
    const int e    = blockIdx.x * 8 + (threadIdx.x >> 5);
    const int row  = ei[e];               // wave-uniform -> scalar loads
    const int col  = ei[E_EDGES + e];
    const float w  = ew[e];
    const float4 hv = *(const float4*)(hin + (size_t)col * HID + lane * 4);
    float* dst = hout + (size_t)row * HID + lane * 4;
    unsafeAtomicAdd(dst + 0, w * hv.x);
    unsafeAtomicAdd(dst + 1, w * hv.y);
    unsafeAtomicAdd(dst + 2, w * hv.z);
    unsafeAtomicAdd(dst + 3, w * hv.w);
}

__global__ void __launch_bounds__(256)
zero_kernel(float4* __restrict__ p, int n4) {
    int i = blockIdx.x * 256 + threadIdx.x;
    if (i < n4) p[i] = make_float4(0.f, 0.f, 0.f, 0.f);
}

__global__ void __launch_bounds__(256)
axpy_kernel(const float* __restrict__ fW, int idx,
            const float4* __restrict__ hnew, float4* __restrict__ hidden, int n4) {
    int i = blockIdx.x * 256 + threadIdx.x;
    if (i >= n4) return;
    const float s = fW[idx];
    float4 a = hnew[i];
    float4 b = hidden[i];
    b.x += s * a.x; b.y += s * a.y; b.z += s * a.z; b.w += s * a.w;
    hidden[i] = b;
}

// ---------------------------------------------------------------------------
// z (+)= hidden @ W_out[branch*128:(branch+1)*128] (+ b_out on branch 0)
// ---------------------------------------------------------------------------
__global__ void __launch_bounds__(256)
out_gemm_kernel(const float* __restrict__ hidden, const float* __restrict__ W_out,
                const float* __restrict__ b_out, float* __restrict__ z, int branch) {
    __shared__ float lw[HID * NCLS];
    for (int i = threadIdx.x; i < HID * NCLS; i += 256)
        lw[i] = W_out[(size_t)(branch * HID + i / NCLS) * NCLS + (i % NCLS)];
    __syncthreads();
    const int node = blockIdx.x * 256 + threadIdx.x;
    if (node >= N_NODES) return;
    float acc[NCLS];
#pragma unroll
    for (int c = 0; c < NCLS; ++c) acc[c] = 0.f;
    const float4* hrow = (const float4*)(hidden + (size_t)node * HID);
    for (int jv = 0; jv < HID / 4; ++jv) {
        float4 h4 = hrow[jv];
        const float* w0 = &lw[(jv * 4) * NCLS];
#pragma unroll
        for (int c = 0; c < NCLS; ++c)
            acc[c] += h4.x * w0[c] + h4.y * w0[NCLS + c]
                    + h4.z * w0[2 * NCLS + c] + h4.w * w0[3 * NCLS + c];
    }
    float* zrow = z + (size_t)node * NCLS;
    if (branch == 0) {
#pragma unroll
        for (int c = 0; c < NCLS; ++c) zrow[c] = b_out[c] + acc[c];
    } else {
#pragma unroll
        for (int c = 0; c < NCLS; ++c) zrow[c] += acc[c];
    }
}

__global__ void __launch_bounds__(256)
lsm_kernel(const float* __restrict__ z, float* __restrict__ out) {
    const int node = blockIdx.x * 256 + threadIdx.x;
    if (node >= N_NODES) return;
    const float* zr = z + (size_t)node * NCLS;
    float zv[NCLS];
#pragma unroll
    for (int c = 0; c < NCLS; ++c) zv[c] = zr[c];
    float m = zv[0];
#pragma unroll
    for (int c = 1; c < NCLS; ++c) m = fmaxf(m, zv[c]);
    float s = 0.f;
#pragma unroll
    for (int c = 0; c < NCLS; ++c) s += expf(zv[c] - m);
    const float ls = logf(s);
    float* o = out + (size_t)node * NCLS;
#pragma unroll
    for (int c = 0; c < NCLS; ++c) o[c] = zv[c] - m - ls;
}

// ---------------------------------------------------------------------------
extern "C" void kernel_launch(void* const* d_in, const int* in_sizes, int n_in,
                              void* d_out, int out_size, void* d_ws, size_t ws_size,
                              hipStream_t stream) {
    const float* x     = (const float*)d_in[0];
    const int*   ei1   = (const int*)  d_in[1];
    const float* ew1   = (const float*)d_in[2];
    const int*   ei2   = (const int*)  d_in[3];
    const float* ew2   = (const float*)d_in[4];
    const float* W_in  = (const float*)d_in[5];
    const float* b_in  = (const float*)d_in[6];
    const float* fW    = (const float*)d_in[7];
    const float* W_out = (const float*)d_in[8];
    const float* b_out = (const float*)d_in[9];
    float* out = (float*)d_out;

    char* ws = (char*)d_ws;
    size_t off = 0;
    auto wsalloc = [&](size_t bytes) -> void* {
        void* p = ws + off;
        off = (off + bytes + 255) & ~(size_t)255;
        return p;
    };
    __bf16* wpack  = (__bf16*)wsalloc((size_t)2 * 16 * 8 * 32 * 16 * sizeof(__bf16));
    float*  hA     = (float*) wsalloc((size_t)N_NODES * HID * sizeof(float));
    float*  hB     = (float*) wsalloc((size_t)N_NODES * HID * sizeof(float));
    float*  hidden = (float*) wsalloc((size_t)N_NODES * HID * sizeof(float));
    float*  z      = (float*) wsalloc((size_t)N_NODES * NCLS * sizeof(float));

    const int n4 = N_NODES * HID / 4;                // 3,200,000
    const int n4blocks = (n4 + 255) / 256;
    const int nodeblocks = (N_NODES + 255) / 256;

    pack_w_kernel<<<512, 256, 0, stream>>>(W_in, wpack);

    for (int br = 0; br < 2; ++br) {
        const int*   ei = br ? ei2 : ei1;
        const float* ew = br ? ew2 : ew1;
        float* hcur = hA;
        float* hnext = hB;

        gemm_in_kernel<<<N_NODES / 16, 256, 0, stream>>>(x, wpack, b_in, fW, br,
                                                         hcur, hidden);
        for (int k = 0; k < KHOPS; ++k) {
            zero_kernel<<<n4blocks, 256, 0, stream>>>((float4*)hnext, n4);
            spmm_kernel<<<E_EDGES / 8, 256, 0, stream>>>(ei, ew, hcur, hnext);
            axpy_kernel<<<n4blocks, 256, 0, stream>>>(fW, br * (KHOPS + 1) + k + 1,
                                                      (const float4*)hnext,
                                                      (float4*)hidden, n4);
            float* t = hcur; hcur = hnext; hnext = t;
        }
        out_gemm_kernel<<<nodeblocks, 256, 0, stream>>>(hidden, W_out, b_out, z, br);
    }
    lsm_kernel<<<nodeblocks, 256, 0, stream>>>(z, out);
}
